// MiniBlock_37288906064321
// MI455X (gfx1250) — compile-verified, matrix operand order checked by reading, not run on hardware
//
#include <hip/hip_runtime.h>

#define DIMN 1024
#define CTRLN 4096
#define HEADSN 16
#define HDN 64
#define BN 2
#define TN 2048
#define BTN (BN*TN)

typedef __bf16 bf16;
typedef __bf16 v16bf __attribute__((ext_vector_type(16)));
typedef float  v8f   __attribute__((ext_vector_type(8)));

__device__ __forceinline__ bf16 f2bf(float f) {
    unsigned u = __builtin_bit_cast(unsigned, f);
    unsigned r = (u + 0x7fffu + ((u >> 16) & 1u)) >> 16;
    unsigned short s = (unsigned short)r;
    return __builtin_bit_cast(bf16, s);
}

// Low 32 bits of a generic pointer to LDS == wave-relative LDS byte address
// (flat aperture mapping, ISA 10.2) — what GLOBAL_LOAD_ASYNC_TO_LDS expects in VDST.
__device__ __forceinline__ unsigned lds_addr(const void* p) {
    return (unsigned)(unsigned long long)p;
}

__device__ __forceinline__ void async_load_b128(unsigned lds, const void* gaddr) {
    asm volatile("global_load_async_to_lds_b128 %0, %1, off"
                 :: "v"(lds), "v"((unsigned long long)gaddr) : "memory");
}

__device__ __forceinline__ void wait_async0() {
    asm volatile("s_wait_asynccnt 0" ::: "memory");
}

// Load a 16x32 bf16 A/B fragment (row = lane&15, ISA 7.12.2 K-pair layout).
// strideHalves must be even; base must be 4-byte aligned.
__device__ __forceinline__ v16bf load_frag(const bf16* base, int lane, int strideHalves) {
    union { v16bf v; unsigned u[8]; } f;
    const unsigned* p = (const unsigned*)(base + (size_t)(lane & 15) * strideHalves);
    int hi4 = ((lane >> 4) & 1) ? 4 : 0;
#pragma unroll
    for (int r = 0; r < 8; ++r)
        f.u[r] = p[r + (r & 4) + hi4];
    return f.v;
}

__device__ __forceinline__ v8f wmma_bf16(v16bf a, v16bf b, v8f c) {
    return __builtin_amdgcn_wmma_f32_16x16x32_bf16(false, a, false, b, (short)0, c, false, false);
}

// ---------------- gates: mass/decay = sigmoid(x . w) ----------------
__global__ void k_gates(const float* __restrict__ x, const float* __restrict__ mw,
                        const float* __restrict__ dw, float* __restrict__ mass,
                        float* __restrict__ decay) {
    int row = blockIdx.x, tid = threadIdx.x;
    const float* xr = x + (size_t)row * DIMN;
    float sm = 0.f, sd = 0.f;
    for (int i = tid; i < DIMN; i += 256) { float v = xr[i]; sm += v * mw[i]; sd += v * dw[i]; }
    __shared__ float r1[256], r2[256];
    r1[tid] = sm; r2[tid] = sd; __syncthreads();
    for (int s = 128; s > 0; s >>= 1) {
        if (tid < s) { r1[tid] += r1[tid + s]; r2[tid] += r2[tid + s]; }
        __syncthreads();
    }
    if (tid == 0) {
        mass[row]  = 1.f / (1.f + __expf(-r1[0]));
        decay[row] = 1.f / (1.f + __expf(-r2[0]));
    }
}

// ---------------- sequential memory scan; xsum = x + mem ----------------
__global__ void k_scan(const float* __restrict__ x, const float* __restrict__ mass,
                       const float* __restrict__ decay, float* __restrict__ xsum) {
    int d = blockIdx.x * blockDim.x + threadIdx.x;
    int b = blockIdx.y;
    float state = 0.f;
    for (int t = 0; t < TN; ++t) {
        int row = b * TN + t;
        size_t idx = (size_t)row * DIMN + d;
        float xv = x[idx];
        state = decay[row] * state + mass[row] * xv;
        xsum[idx] = xv + state;
    }
}

// ---------------- LN1 + sign-folded bf16 Q/K/V ----------------
__global__ void k_ln1(const float* __restrict__ xs, const float* __restrict__ g,
                      const float* __restrict__ bt, const float* __restrict__ bq,
                      const float* __restrict__ bk, const float* __restrict__ bv,
                      float* __restrict__ x1, bf16* __restrict__ Q,
                      bf16* __restrict__ K, bf16* __restrict__ V) {
    int row = blockIdx.x, tid = threadIdx.x;
    size_t base = (size_t)row * DIMN;
    float vals[4], s = 0.f, ss = 0.f;
#pragma unroll
    for (int i = 0; i < 4; ++i) { float v = xs[base + tid + i * 256]; vals[i] = v; s += v; ss += v * v; }
    __shared__ float r1[256], r2[256];
    r1[tid] = s; r2[tid] = ss; __syncthreads();
    for (int st = 128; st > 0; st >>= 1) {
        if (tid < st) { r1[tid] += r1[tid + st]; r2[tid] += r2[tid + st]; }
        __syncthreads();
    }
    float mean = r1[0] * (1.f / DIMN);
    float var  = r2[0] * (1.f / DIMN) - mean * mean;
    float inv  = rsqrtf(var + 1e-5f);
#pragma unroll
    for (int i = 0; i < 4; ++i) {
        int d = tid + i * 256;
        float y = (vals[i] - mean) * inv * g[d] + bt[d];
        size_t idx = base + d;
        x1[idx] = y;
        float sq = (bq[d] > 0.f) ? 1.f : ((bq[d] < 0.f) ? -1.f : 0.f);
        float sk = (bk[d] > 0.f) ? 1.f : ((bk[d] < 0.f) ? -1.f : 0.f);
        float sv = (bv[d] > 0.f) ? 1.f : ((bv[d] < 0.f) ? -1.f : 0.f);
        Q[idx] = f2bf(y * sq); K[idx] = f2bf(y * sk); V[idx] = f2bf(y * sv);
    }
}

// ---------------- causal sigmoid attention, WMMA bf16 ----------------
__global__ void __launch_bounds__(256) k_attn(const bf16* __restrict__ Q,
                                              const bf16* __restrict__ K,
                                              const bf16* __restrict__ V,
                                              float* __restrict__ attn) {
    int tblk = blockIdx.x * 128;
    int h = blockIdx.y, b = blockIdx.z;
    int tid = threadIdx.x, lane = tid & 31, wave = tid >> 5;
    int tw0 = tblk + wave * 16;

    __shared__ bf16 Ks[32 * 64];     // [s][hd]
    __shared__ bf16 Vt[64 * 32];     // [hd][s]
    __shared__ bf16 Sst[8][16 * 32]; // per-wave score staging

    const bf16* Qb = Q + ((size_t)b * TN) * DIMN + h * HDN;
    const bf16* Kb = K + ((size_t)b * TN) * DIMN + h * HDN;
    const bf16* Vb = V + ((size_t)b * TN) * DIMN + h * HDN;

    v16bf qa0 = load_frag(Qb + (size_t)tw0 * DIMN,      lane, DIMN); // hd 0..31
    v16bf qa1 = load_frag(Qb + (size_t)tw0 * DIMN + 32, lane, DIMN); // hd 32..63

    v8f zero = {};
    v8f acc[4]; acc[0] = zero; acc[1] = zero; acc[2] = zero; acc[3] = zero;
    const float scale4 = 4.0f * 0.125f; // 4 * HD^-0.5

    unsigned ks_base = lds_addr(&Ks[0]);
    int idx8 = tid * 8;                 // 2048 halves / 256 threads
    int sload = idx8 >> 6, hdload = idx8 & 63;

    int smax = tblk + 128;
    for (int sb = 0; sb < smax; sb += 32) {
        // K tile: 128-bit async copy straight into LDS (row-major, contiguous)
        async_load_b128(ks_base + (unsigned)idx8 * 2,
                        Kb + (size_t)(sb + sload) * DIMN + hdload);
        // V tile: 128-bit global load, transpose via 8 x ds_store_b16
        union { uint4 q; unsigned short hw[8]; } uv;
        uv.q = *(const uint4*)(Vb + (size_t)(sb + sload) * DIMN + hdload);
#pragma unroll
        for (int j = 0; j < 8; ++j)
            Vt[(hdload + j) * 32 + sload] = __builtin_bit_cast(bf16, uv.hw[j]);
        wait_async0();
        __syncthreads();
        if (sb <= tw0 + 15) {
            int n = lane & 15, hi = (lane >> 4) & 1;
#pragma unroll
            for (int nt = 0; nt < 2; ++nt) {
                v8f sacc = zero;
                v16bf b0 = load_frag(Ks + nt * 16 * 64,      lane, 64);
                v16bf b1 = load_frag(Ks + nt * 16 * 64 + 32, lane, 64);
                sacc = wmma_bf16(qa0, b0, sacc);
                sacc = wmma_bf16(qa1, b1, sacc);
                int scol = sb + nt * 16 + n;
#pragma unroll
                for (int r = 0; r < 8; ++r) {
                    int trow = tw0 + r + hi * 8;
                    float v = sacc[r];
                    v = (scol <= trow) ? (1.f / (1.f + __expf(-scale4 * v))) : 0.f;
                    Sst[wave][(r + hi * 8) * 32 + nt * 16 + n] = f2bf(v);
                }
            }
            asm volatile("s_wait_dscnt 0" ::: "memory");
            v16bf sa = load_frag(&Sst[wave][0], lane, 32);
#pragma unroll
            for (int nt = 0; nt < 4; ++nt) {
                v16bf vb = load_frag(Vt + nt * 16 * 32, lane, 32);
                acc[nt] = wmma_bf16(sa, vb, acc[nt]);
            }
        }
        __syncthreads();
    }
    int n = lane & 15, hi = (lane >> 4) & 1;
    float* ab = attn + ((size_t)b * TN) * DIMN + h * HDN;
#pragma unroll
    for (int nt = 0; nt < 4; ++nt)
#pragma unroll
        for (int r = 0; r < 8; ++r)
            ab[(size_t)(tw0 + r + hi * 8) * DIMN + nt * 16 + n] = acc[nt][r];
}

// ---------------- LN2 (residual) fused with controller-LN -> bf16 ----------------
__global__ void k_ln2(const float* __restrict__ x1, const float* __restrict__ at,
                      const float* __restrict__ g2, const float* __restrict__ b2,
                      const float* __restrict__ gc, const float* __restrict__ bc,
                      float* __restrict__ x2, bf16* __restrict__ hc) {
    int row = blockIdx.x, tid = threadIdx.x;
    size_t base = (size_t)row * DIMN;
    __shared__ float r1[256], r2[256];
    float v[4], s = 0.f, ss = 0.f;
#pragma unroll
    for (int i = 0; i < 4; ++i) {
        float z = x1[base + tid + i * 256] + at[base + tid + i * 256];
        v[i] = z; s += z; ss += z * z;
    }
    r1[tid] = s; r2[tid] = ss; __syncthreads();
    for (int st = 128; st > 0; st >>= 1) {
        if (tid < st) { r1[tid] += r1[tid + st]; r2[tid] += r2[tid + st]; }
        __syncthreads();
    }
    float mean = r1[0] * (1.f / DIMN);
    float inv  = rsqrtf(r2[0] * (1.f / DIMN) - mean * mean + 1e-5f);
    float y[4], s2 = 0.f, ss2 = 0.f;
#pragma unroll
    for (int i = 0; i < 4; ++i) {
        int d = tid + i * 256;
        y[i] = (v[i] - mean) * inv * g2[d] + b2[d];
        x2[base + d] = y[i];
        s2 += y[i]; ss2 += y[i] * y[i];
    }
    __syncthreads();
    r1[tid] = s2; r2[tid] = ss2; __syncthreads();
    for (int st = 128; st > 0; st >>= 1) {
        if (tid < st) { r1[tid] += r1[tid + st]; r2[tid] += r2[tid + st]; }
        __syncthreads();
    }
    float mean2 = r1[0] * (1.f / DIMN);
    float inv2  = rsqrtf(r2[0] * (1.f / DIMN) - mean2 * mean2 + 1e-5f);
#pragma unroll
    for (int i = 0; i < 4; ++i) {
        int d = tid + i * 256;
        hc[base + d] = f2bf((y[i] - mean2) * inv2 * gc[d] + bc[d]);
    }
}

// ---------------- f32 -> bf16 weight conversion ----------------
__global__ void k_f2bf(const float* __restrict__ in, bf16* __restrict__ out, int n) {
    int i = blockIdx.x * 256 + threadIdx.x;
    if (i < n) out[i] = f2bf(in[i]);
}

// ---------------- tiled WMMA GEMM: mode 0 = gelu->bf16, mode 1 = +bias+res->f32 ----
template <int MODE>
__global__ void __launch_bounds__(256) k_gemm(const bf16* __restrict__ A,
                                              const bf16* __restrict__ B,
                                              const float* __restrict__ bias,
                                              const float* __restrict__ res,
                                              bf16* __restrict__ obf,
                                              float* __restrict__ of,
                                              int M, int N, int Kd) {
    int m0 = blockIdx.y * 128;
    int n0 = blockIdx.x * 64;
    int tid = threadIdx.x, lane = tid & 31, wave = tid >> 5;
    __shared__ bf16 As[128 * 32]; // [m][k]  (contiguous image of the global tile)
    __shared__ bf16 Bt[64 * 32];  // [n][k]  (transposed)
    v8f zero = {};
    v8f acc[4]; acc[0] = zero; acc[1] = zero; acc[2] = zero; acc[3] = zero;
    unsigned as_base = lds_addr(&As[0]);
    int idx8 = tid * 8;                   // B-tile: 2048 halves / 256 threads
    int kB = idx8 >> 6, nB = idx8 & 63;

    for (int kb = 0; kb < Kd; kb += 32) {
        // A tile: 512 x 16B segments, async global->LDS (layout-preserving)
#pragma unroll
        for (int i = 0; i < 2; ++i) {
            int seg = tid + i * 256;          // 0..511
            int row = seg >> 2, off8 = (seg & 3) * 8;
            async_load_b128(as_base + (unsigned)seg * 16,
                            A + (size_t)(m0 + row) * Kd + kb + off8);
        }
        // B tile: 128-bit global load + transpose store
        union { uint4 q; unsigned short hw[8]; } ub;
        ub.q = *(const uint4*)(B + (size_t)(kb + kB) * N + n0 + nB);
#pragma unroll
        for (int j = 0; j < 8; ++j)
            Bt[(nB + j) * 32 + kB] = __builtin_bit_cast(bf16, ub.hw[j]);
        wait_async0();
        __syncthreads();
        v16bf a = load_frag(As + wave * 16 * 32, lane, 32);
#pragma unroll
        for (int nt = 0; nt < 4; ++nt) {
            v16bf bfrag = load_frag(Bt + nt * 16 * 32, lane, 32);
            acc[nt] = wmma_bf16(a, bfrag, acc[nt]);
        }
        __syncthreads();
    }

    int n = lane & 15, hi = (lane >> 4) & 1;
#pragma unroll
    for (int nt = 0; nt < 4; ++nt) {
#pragma unroll
        for (int r = 0; r < 8; ++r) {
            int gm = m0 + wave * 16 + r + hi * 8;
            int gn = n0 + nt * 16 + n;
            size_t oi = (size_t)gm * N + gn;
            float v = acc[nt][r] + bias[gn];
            if (MODE == 0) {
                v = 0.5f * v * (1.0f + erff(v * 0.70710678118f)); // exact gelu
                obf[oi] = f2bf(v);
            } else {
                of[oi] = v + res[oi];
            }
        }
    }
}

extern "C" void kernel_launch(void* const* d_in, const int* in_sizes, int n_in,
                              void* d_out, int out_size, void* d_ws, size_t ws_size,
                              hipStream_t stream) {
    const float* x      = (const float*)d_in[0];
    const float* mass_w = (const float*)d_in[1];
    const float* decay_w= (const float*)d_in[2];
    const float* bv_q   = (const float*)d_in[3];
    const float* bv_k   = (const float*)d_in[4];
    const float* bv_v   = (const float*)d_in[5];
    const float* ln1g   = (const float*)d_in[6];
    const float* ln1b   = (const float*)d_in[7];
    const float* ln2g   = (const float*)d_in[8];
    const float* ln2b   = (const float*)d_in[9];
    const float* lncg   = (const float*)d_in[10];
    const float* lncb   = (const float*)d_in[11];
    const float* w_down = (const float*)d_in[12];
    const float* b_down = (const float*)d_in[13];
    const float* w_up   = (const float*)d_in[14];
    const float* b_up   = (const float*)d_in[15];
    float* out = (float*)d_out;

    char* ws = (char*)d_ws;
    size_t off = 0;
    auto alloc = [&](size_t bytes) -> void* {
        off = (off + 255) & ~(size_t)255;
        void* p = ws + off; off += bytes; return p;
    };
    const size_t BTD = (size_t)BTN * DIMN;
    float* mass  = (float*)alloc(BTN * 4);
    float* decay = (float*)alloc(BTN * 4);
    float* xsum  = (float*)alloc(BTD * 4);   // later: attn output, then lower half of hmid
    float* x1    = (float*)alloc(BTD * 4);   // later: upper half of hmid
    bf16*  Qb    = (bf16*)alloc(BTD * 2);
    bf16*  Kb    = (bf16*)alloc(BTD * 2);
    bf16*  Vb    = (bf16*)alloc(BTD * 2);
    float* x2    = (float*)alloc(BTD * 4);
    bf16*  hc    = (bf16*)alloc(BTD * 2);
    bf16*  wdb   = (bf16*)alloc((size_t)DIMN * CTRLN * 2);
    bf16*  wub   = (bf16*)alloc((size_t)CTRLN * DIMN * 2);
    float* attn  = xsum;                     // alias: xsum dead after LN1
    bf16*  hmid  = (bf16*)xsum;              // alias: xsum+x1 (32 MiB) dead after LN2

    // 0) weights -> bf16
    int nw = DIMN * CTRLN;
    k_f2bf<<<(nw + 255) / 256, 256, 0, stream>>>(w_down, wdb, nw);
    k_f2bf<<<(nw + 255) / 256, 256, 0, stream>>>(w_up,   wub, nw);
    // 1) gates
    k_gates<<<BTN, 256, 0, stream>>>(x, mass_w, decay_w, mass, decay);
    // 2) sequential memory scan -> xsum = x + mem
    k_scan<<<dim3(DIMN / 256, BN), 256, 0, stream>>>(x, mass, decay, xsum);
    // 3) LN1 + Q/K/V bf16
    k_ln1<<<BTN, 256, 0, stream>>>(xsum, ln1g, ln1b, bv_q, bv_k, bv_v, x1, Qb, Kb, Vb);
    // 4) attention (WMMA) -> attn (aliases xsum)
    k_attn<<<dim3(TN / 128, HEADSN, BN), 256, 0, stream>>>(Qb, Kb, Vb, attn);
    // 5) LN2 + controller LN -> x2 (f32), hc (bf16)
    k_ln2<<<BTN, 256, 0, stream>>>(x1, attn, ln2g, ln2b, lncg, lncb, x2, hc);
    // 6) GEMM1: hmid = gelu(hc @ w_down + b_down)   [BT x CTRL]
    k_gemm<0><<<dim3(CTRLN / 64, BTN / 128), 256, 0, stream>>>(
        hc, wdb, b_down, nullptr, hmid, nullptr, BTN, CTRLN, DIMN);
    // 7) GEMM2: out = x2 + hmid @ w_up + b_up       [BT x DIM]
    k_gemm<1><<<dim3(DIMN / 64, BTN / 128), 256, 0, stream>>>(
        hmid, wub, b_up, x2, nullptr, out, BTN, DIMN, CTRLN);
}
